// PositionWiseFCNetwork_31172872635042
// MI455X (gfx1250) — matrix operand order repaired and checked by reading, not compiled, not run on hardware
//
#include <hip/hip_runtime.h>
#include <hip/hip_bf16.h>

// ---------------- problem constants ----------------
#define T_TOK 16384          // N*P tokens
#define DD    1024           // model dim
#define HH    4096           // hidden dim
#define EE    8              // experts
#define KCE   256            // K-chunk (expert GEMM, double-buffered async)
#define KCC   512            // K-chunk (condense GEMM, fused combine staging)
#define LIST_CAP (2 * T_TOK) // per-expert worst-case entries
#define TRASH_ROW (2 * T_TOK)

typedef __attribute__((ext_vector_type(16))) __bf16 v16bf;
typedef __attribute__((ext_vector_type(8)))  __bf16 v8bf;
typedef __attribute__((ext_vector_type(4)))  __bf16 v4bf;
typedef __attribute__((ext_vector_type(8)))  float  v8f;
typedef __attribute__((ext_vector_type(4)))  int    v4i;

// -------- CDNA5 async global->LDS path (ASYNCcnt), with safe fallback --------
#if defined(__has_builtin)
#if __has_builtin(__builtin_amdgcn_global_load_async_to_lds_b128) && \
    __has_builtin(__builtin_amdgcn_s_wait_asynccnt)
#define USE_ASYNC_LDS 1
#endif
#endif

typedef __attribute__((address_space(1))) v4i as1_v4i;  // prints as "__device__" in diags
typedef __attribute__((address_space(3))) v4i as3_v4i;

__device__ inline void stage_b128(const __bf16* gsrc, __bf16* ldst) {
#if defined(USE_ASYNC_LDS)
  // per-lane 16B global -> LDS, tracked by ASYNCcnt (no VGPR transit)
  __builtin_amdgcn_global_load_async_to_lds_b128(
      (as1_v4i*)gsrc, (as3_v4i*)ldst, 0, 0);
#else
  *(uint4*)ldst = *(const uint4*)gsrc;
#endif
}

__device__ inline void wait_async_stage() {
#if defined(USE_ASYNC_LDS)
  __builtin_amdgcn_s_wait_asynccnt(0);
#endif
}

// ---------------- WMMA helpers ----------------
__device__ inline v8f wmma_bf16(v16bf a, v16bf b, v8f c) {
  // D = A(16x32 bf16) * B(32x16 bf16) + C(16x16 f32)
  return __builtin_amdgcn_wmma_f32_16x16x32_bf16(false, a, false, b,
                                                 (short)0, c, false, false);
}

// A fragment (16x32 bf16) from LDS, rows row0..row0+15, K offset dOff.
// Contiguous K order used here: lanes 0-15 -> M=lane, K=0..15; lanes 16-31 -> K=16..31 (matches B fragment).
__device__ inline v16bf a_frag_lds(const __bf16* sA, int rowStride, int row0, int dOff) {
  int lane = threadIdx.x & 31;
  int m = lane & 15;
  int bk = (lane < 16) ? 0 : 16;
  const __bf16* p = sA + (size_t)(row0 + m) * rowStride + dOff + bk;
  v8bf lo = *(const v8bf*)p;          // K = bk .. bk+7
  v8bf hi = *(const v8bf*)(p + 8);    // K = bk+8 .. bk+15
  return __builtin_shufflevector(lo, hi, 0,1,2,3,4,5,6,7,8,9,10,11,12,13,14,15);
}

// B fragment (32x16 bf16) from global, column col0+n, K contiguous.
// Contiguous K order used here: lanes 0-15 -> N=lane, K=0..15; lanes 16-31 -> N=lane-16, K=16..31.
__device__ inline v16bf b_frag_global(const __bf16* W, int ldw, int col0, int k0) {
  int lane = threadIdx.x & 31;
  int n = lane & 15;
  int koff = (lane < 16) ? 0 : 16;
  return *(const v16bf*)(W + (size_t)(col0 + n) * ldw + k0 + koff);
}

// ---------------- init workspace ----------------
__global__ __launch_bounds__(256) void moe_init_ws(int* __restrict__ lists,
                                                   int* __restrict__ cnt,
                                                   __bf16* __restrict__ xbf) {
  int i = blockIdx.x * 256 + threadIdx.x;
  if (i < EE * LIST_CAP) lists[i] = TRASH_ROW;      // padded entries -> trash row
  if (i < EE) cnt[i] = 0;
  if (i < DD) xbf[(size_t)T_TOK * DD + i] = (__bf16)0.0f;  // trash x row
}

// ---------------- f32 -> bf16 weight conversion ----------------
__global__ __launch_bounds__(256) void moe_cvt_bf16(const float* __restrict__ src,
                                                    __bf16* __restrict__ dst, int n4) {
  int i = blockIdx.x * 256 + threadIdx.x;
  if (i < n4) {
    float4 v = ((const float4*)src)[i];
    v4bf o;
    o[0] = (__bf16)v.x; o[1] = (__bf16)v.y; o[2] = (__bf16)v.z; o[3] = (__bf16)v.w;
    ((v4bf*)dst)[i] = o;
  }
}

// ---------------- LayerNorm + gating + routing ----------------
__global__ __launch_bounds__(256) void moe_ln_gate(
    const float* __restrict__ seq, const float* __restrict__ gamma,
    const float* __restrict__ beta, const float* __restrict__ Wg,
    const float* __restrict__ bg, __bf16* __restrict__ xbf,
    int* __restrict__ lists, int* __restrict__ cnt) {
  int t = blockIdx.x;
  int tid = threadIdx.x;
  int d = tid * 4;
  float4 v = *(const float4*)(seq + (size_t)t * DD + d);
  float s = v.x + v.y + v.z + v.w;
  float q = v.x * v.x + v.y * v.y + v.z * v.z + v.w * v.w;
  for (int off = 16; off > 0; off >>= 1) {
    s += __shfl_down(s, off, 32);
    q += __shfl_down(q, off, 32);
  }
  __shared__ float rs[8], rq[8], stat[2];
  __shared__ float rl[8][EE];
  int wave = tid >> 5, lane = tid & 31;
  if (lane == 0) { rs[wave] = s; rq[wave] = q; }
  __syncthreads();
  if (tid == 0) {
    float S = 0.f, Q = 0.f;
    for (int i = 0; i < 8; ++i) { S += rs[i]; Q += rq[i]; }
    float mu = S * (1.0f / DD);
    float var = Q * (1.0f / DD) - mu * mu;
    stat[0] = mu;
    stat[1] = rsqrtf(var + 1e-5f);
  }
  __syncthreads();
  float mu = stat[0], rsig = stat[1];
  float n0 = (v.x - mu) * rsig * gamma[d + 0] + beta[d + 0];
  float n1 = (v.y - mu) * rsig * gamma[d + 1] + beta[d + 1];
  float n2 = (v.z - mu) * rsig * gamma[d + 2] + beta[d + 2];
  float n3 = (v.w - mu) * rsig * gamma[d + 3] + beta[d + 3];
  v4bf nb;
  nb[0] = (__bf16)n0; nb[1] = (__bf16)n1; nb[2] = (__bf16)n2; nb[3] = (__bf16)n3;
  *(v4bf*)(xbf + (size_t)t * DD + d) = nb;
  // gating partial dot products (softmax is monotonic -> rank raw logits)
  float p[EE];
#pragma unroll
  for (int e = 0; e < EE; ++e) {
    const float* wr = Wg + e * DD + d;
    p[e] = n0 * wr[0] + n1 * wr[1] + n2 * wr[2] + n3 * wr[3];
  }
#pragma unroll
  for (int e = 0; e < EE; ++e)
    for (int off = 16; off > 0; off >>= 1) p[e] += __shfl_down(p[e], off, 32);
  if (lane == 0)
    for (int e = 0; e < EE; ++e) rl[wave][e] = p[e];
  __syncthreads();
  if (tid == 0) {
    float lg[EE];
    for (int e = 0; e < EE; ++e) {
      float acc = bg[e];
      for (int w = 0; w < 8; ++w) acc += rl[w][e];
      lg[e] = acc;
    }
    int e0 = 0;
    for (int e = 1; e < EE; ++e) if (lg[e] > lg[e0]) e0 = e;
    int e1 = (e0 == 0) ? 1 : 0;
    for (int e = 0; e < EE; ++e) if (e != e0 && lg[e] > lg[e1]) e1 = e;
    int p0 = atomicAdd(&cnt[e0], 1);
    lists[e0 * LIST_CAP + p0] = t * 2 + 0;   // hidden row = 2t + slot
    int p1 = atomicAdd(&cnt[e1], 1);
    lists[e1 * LIST_CAP + p1] = t * 2 + 1;
  }
}

// ---------------- expert GEMM: hs[row] = x @ We[e]^T + be[e] ----------------
// grid.x = E * 1024 token tiles (32 tokens each, early exit), grid.y = H/256
// Double-buffered LDS A staging via async global->LDS (ASYNCcnt).
__global__ __launch_bounds__(256) void moe_expert_gemm(
    const __bf16* __restrict__ xbf, const __bf16* __restrict__ Webf,
    const float* __restrict__ be, const int* __restrict__ lists,
    const int* __restrict__ cnt, __bf16* __restrict__ hs) {
  __shared__ __align__(16) __bf16 sA[2][32 * KCE];
  __shared__ int sRow[32];
  int e = blockIdx.x >> 10;
  int tile = blockIdx.x & 1023;
  int c = cnt[e];
  if (tile * 32 >= c) return;
  if (threadIdx.x < 32) sRow[threadIdx.x] = lists[e * LIST_CAP + tile * 32 + threadIdx.x];
  __syncthreads();

  int wave = threadIdx.x >> 5, lane = threadIdx.x & 31;
  int n0 = blockIdx.y * 256 + wave * 32;      // two 16-wide N tiles: n0, n0+16
  const __bf16* Wbase = Webf + (size_t)e * HH * DD;
  v8f acc00 = {}, acc01 = {}, acc10 = {}, acc11 = {};

  int crow = threadIdx.x >> 3;                // staging: 8 threads per row
  int ccol = (threadIdx.x & 7) * 32;          // 32 halves (4 x b128) per thread
  const __bf16* grow = xbf + (size_t)(sRow[crow] >> 1) * DD + ccol;

  // prologue: stage chunk 0 into buffer 0
#pragma unroll
  for (int i = 0; i < 4; ++i)
    stage_b128(grow + i * 8, &sA[0][crow * KCE + ccol + i * 8]);
  wait_async_stage();
  __syncthreads();

  int bufc = 0;
  for (int kc = 0; kc < DD; kc += KCE) {
    // issue async fill of the next chunk while this chunk feeds the WMMAs
    if (kc + KCE < DD) {
#pragma unroll
      for (int i = 0; i < 4; ++i)
        stage_b128(grow + kc + KCE + i * 8,
                   &sA[bufc ^ 1][crow * KCE + ccol + i * 8]);
    }
    const __bf16* cur = sA[bufc];
#pragma unroll 4
    for (int d0 = 0; d0 < KCE; d0 += 32) {
      v16bf a0 = a_frag_lds(cur, KCE, 0, d0);
      v16bf a1 = a_frag_lds(cur, KCE, 16, d0);
      v16bf b0 = b_frag_global(Wbase, DD, n0, kc + d0);
      v16bf b1 = b_frag_global(Wbase, DD, n0 + 16, kc + d0);
      acc00 = wmma_bf16(a0, b0, acc00);
      acc01 = wmma_bf16(a0, b1, acc01);
      acc10 = wmma_bf16(a1, b0, acc10);
      acc11 = wmma_bf16(a1, b1, acc11);
    }
    wait_async_stage();
    __syncthreads();
    bufc ^= 1;
  }

  int ncol = lane & 15;
  int rbase = (lane >> 4) * 8;
  float bias0 = be[e * HH + n0 + ncol];
  float bias1 = be[e * HH + n0 + 16 + ncol];
#pragma unroll
  for (int r = 0; r < 8; ++r) {
    int m0 = r + rbase;
    int hr0 = sRow[m0];
    int hr1 = sRow[m0 + 16];
    hs[(size_t)hr0 * HH + n0 + ncol]      = (__bf16)(acc00[r] + bias0);
    hs[(size_t)hr0 * HH + n0 + 16 + ncol] = (__bf16)(acc01[r] + bias1);
    hs[(size_t)hr1 * HH + n0 + ncol]      = (__bf16)(acc10[r] + bias0);
    hs[(size_t)hr1 * HH + n0 + 16 + ncol] = (__bf16)(acc11[r] + bias1);
  }
}

// ---------------- condense: out = x_in + relu((s0+s1)/2) @ Wc^T + bc ----------------
// grid.x = T/32, grid.y = D/256.  Staging fuses slot-combine + relu (VALU),
// so it stays synchronous (data must transit registers anyway).
__global__ __launch_bounds__(256) void moe_condense_gemm(
    const __bf16* __restrict__ hs, const __bf16* __restrict__ Wcbf,
    const float* __restrict__ bc, const float* __restrict__ seq,
    float* __restrict__ out) {
  __shared__ __align__(16) __bf16 sA[32 * KCC];
  int t0 = blockIdx.x * 32;
  int wave = threadIdx.x >> 5, lane = threadIdx.x & 31;
  int n0 = blockIdx.y * 256 + wave * 32;
  v8f acc00 = {}, acc01 = {}, acc10 = {}, acc11 = {};

  int crow = threadIdx.x >> 3;
  int ccol = (threadIdx.x & 7) * 64;
  for (int kc = 0; kc < HH; kc += KCC) {
    const v8bf* s0 = (const v8bf*)(hs + (size_t)(2 * (t0 + crow)) * HH + kc + ccol);
    const v8bf* s1 = (const v8bf*)(hs + (size_t)(2 * (t0 + crow) + 1) * HH + kc + ccol);
    v8bf* dst = (v8bf*)(sA + crow * KCC + ccol);
#pragma unroll
    for (int i = 0; i < 8; ++i) {
      v8bf va = s0[i];
      v8bf vb = s1[i];
      v8bf r;
#pragma unroll
      for (int j = 0; j < 8; ++j) {
        float f = ((float)va[j] + (float)vb[j]) * 0.5f;
        r[j] = (__bf16)fmaxf(f, 0.0f);
      }
      dst[i] = r;
    }
    __syncthreads();
#pragma unroll 4
    for (int d0 = 0; d0 < KCC; d0 += 32) {
      v16bf a0 = a_frag_lds(sA, KCC, 0, d0);
      v16bf a1 = a_frag_lds(sA, KCC, 16, d0);
      v16bf b0 = b_frag_global(Wcbf, HH, n0, kc + d0);
      v16bf b1 = b_frag_global(Wcbf, HH, n0 + 16, kc + d0);
      acc00 = wmma_bf16(a0, b0, acc00);
      acc01 = wmma_bf16(a0, b1, acc01);
      acc10 = wmma_bf16(a1, b0, acc10);
      acc11 = wmma_bf16(a1, b1, acc11);
    }
    __syncthreads();
  }

  int ncol = lane & 15;
  int rbase = (lane >> 4) * 8;
  float b0 = bc[n0 + ncol];
  float b1 = bc[n0 + 16 + ncol];
#pragma unroll
  for (int r = 0; r < 8; ++r) {
    int tokA = t0 + r + rbase;
    int tokB = tokA + 16;
    size_t i00 = (size_t)tokA * DD + n0 + ncol;
    size_t i01 = (size_t)tokA * DD + n0 + 16 + ncol;
    size_t i10 = (size_t)tokB * DD + n0 + ncol;
    size_t i11 = (size_t)tokB * DD + n0 + 16 + ncol;
    out[i00] = acc00[r] + b0 + seq[i00];
    out[i01] = acc01[r] + b1 + seq[i01];
    out[i10] = acc10[r] + b0 + seq[i10];
    out[i11] = acc11[r] + b1 + seq[i11];
  }
}

// ---------------- launch ----------------
extern "C" void kernel_launch(void* const* d_in, const int* in_sizes, int n_in,
                              void* d_out, int out_size, void* d_ws, size_t ws_size,
                              hipStream_t stream) {
  (void)in_sizes; (void)n_in; (void)out_size; (void)ws_size;
  const float* seq   = (const float*)d_in[0];
  const float* gamma = (const float*)d_in[1];
  const float* beta  = (const float*)d_in[2];
  const float* Wg    = (const float*)d_in[3];
  const float* bg    = (const float*)d_in[4];
  const float* We    = (const float*)d_in[5];
  const float* be    = (const float*)d_in[6];
  const float* Wc    = (const float*)d_in[7];
  const float* bc    = (const float*)d_in[8];
  float* out = (float*)d_out;

  // workspace carve-up (all offsets 256B aligned)
  char* ws = (char*)d_ws;
  size_t off = 0;
  __bf16* xbf  = (__bf16*)(ws + off); off += (size_t)(T_TOK + 1) * DD * 2 + 2048;
  off = (off + 255) & ~(size_t)255;
  __bf16* Webf = (__bf16*)(ws + off); off += (size_t)EE * HH * DD * 2;
  __bf16* Wcbf = (__bf16*)(ws + off); off += (size_t)DD * HH * 2;
  __bf16* hs   = (__bf16*)(ws + off); off += (size_t)(2 * T_TOK + 8) * HH * 2;
  int* lists   = (int*)(ws + off);    off += (size_t)EE * LIST_CAP * 4;
  int* cnt     = (int*)(ws + off);    off += 256;

  // 1. init counters / trash entries / trash x row
  moe_init_ws<<<(EE * LIST_CAP) / 256, 256, 0, stream>>>(lists, cnt, xbf);
  // 2. weights -> bf16 (L2-resident thereafter: 64MB + 8MB << 192MB L2)
  moe_cvt_bf16<<<(EE * HH * DD / 4) / 256, 256, 0, stream>>>(We, Webf, EE * HH * DD / 4);
  moe_cvt_bf16<<<(DD * HH / 4) / 256, 256, 0, stream>>>(Wc, Wcbf, DD * HH / 4);
  // 3. LayerNorm + gate + route
  moe_ln_gate<<<T_TOK, 256, 0, stream>>>(seq, gamma, beta, Wg, bg, xbf, lists, cnt);
  // 4. routed expert GEMMs (E * worst-case tiles, early exit) x H tiles
  moe_expert_gemm<<<dim3(EE * 1024, HH / 256), 256, 0, stream>>>(xbf, Webf, be, lists, cnt, hs);
  // 5. fused combine/relu + condense GEMM + bias + residual
  moe_condense_gemm<<<dim3(T_TOK / 32, DD / 256), 256, 0, stream>>>(hs, Wcbf, bc, seq, out);
}
